// Attention_15040975470969
// MI455X (gfx1250) — matrix operand order repaired and checked by reading, not compiled, not run on hardware
//
#include <hip/hip_runtime.h>
#include <hip/hip_bf16.h>
#include <math.h>

// ---------------- problem constants (from reference) ----------------
#define NC    512
#define NB    128
#define NT    256
#define HID   512
#define NCLS1 96          // NCLS-1
#define STEPS 32
#define MROWS (NB*NT)     // 32768 rows of feats_proj

typedef __attribute__((ext_vector_type(16))) _Float16 v16h;
typedef __attribute__((ext_vector_type(8)))  _Float16 v8h;
typedef __attribute__((ext_vector_type(8)))  float    v8f;

// ---------------- WMMA 16x16 tile GEMM (pure f16 operands) ----------
// D(16x16) = A(16xK) * B(Kx16) (+ bias[n]); A,B f16; f32 accumulate.
// A: row-major [M,K], lda = K.  B: weights [N,K] row-major, ldb = K.
//
// Fragment layouts used by this code (wave32, contiguous k order, not the ISA table order):
//  A 16x32 f16: lane L<16 holds row L, K = k0+0..15
//               lane L>=16 holds row L-16, K = k0+16..31
//   -> two contiguous v8h (16B) loads per lane.
//  B 32x16 f16: lanes 0-15: K = k0+0..15 ; lanes 16-31: K = k0+16..31 ; N = lane%16
//   -> one contiguous 32B load per lane.
//  C/D 16x16 f32: VGPR r -> M = r + (lane<16 ? 0 : 8); N = lane%16
__device__ __forceinline__ void gemm_tile_f16wmma(
    const _Float16* __restrict__ A, long lda, long m0,
    const _Float16* __restrict__ B, long ldb, long n0,
    int K,
    float* __restrict__ out, long ldo,
    const float* __restrict__ bias)
{
  const int  lane = threadIdx.x & 31;
  const bool hi   = lane >= 16;
  const int  row  = lane & 15;                 // A-row (half) and B/D column
  const _Float16* ap = A + (m0 + row) * lda + (hi ? 16 : 0);
  const _Float16* bp = B + (n0 + row) * ldb + (hi ? 16 : 0);
  v8f acc = {};
#pragma unroll 4
  for (int k0 = 0; k0 < K; k0 += 32) {
    __builtin_prefetch(ap + k0 + 256, 0, 1);   // speculative global_prefetch_b8
    v8h  a0 = *(const v8h*)(ap + k0);          // K kbase+0..7
    v8h  a1 = *(const v8h*)(ap + k0 + 8);      // K kbase+8..15
    v16h b  = *(const v16h*)(bp + k0);         // K kbase+0..15 (32B)
    v16h a  = __builtin_shufflevector(a0, a1, 0, 1, 2, 3, 4, 5, 6, 7,
                                              8, 9, 10, 11, 12, 13, 14, 15);
    acc = __builtin_amdgcn_wmma_f32_16x16x32_f16(
        /*neg_a=*/false, a, /*neg_b=*/false, b,
        /*c_mod=*/(short)0, acc, /*reuse_a=*/false, /*reuse_b=*/false);
  }
  float bv = bias ? bias[n0 + row] : 0.0f;
  float* op = out + (m0 + (hi ? 8 : 0)) * ldo + n0 + row;
#pragma unroll
  for (int r = 0; r < 8; ++r) {
    op[(long)r * ldo] = acc[r] + bv;
  }
}

// Generic GEMM: linear tile id over (m_tile, n_tile); 8 waves per block.
__global__ void gemm_f16_kernel(const _Float16* __restrict__ A, long lda,
                                const _Float16* __restrict__ B, long ldb,
                                const float* __restrict__ bias,
                                float* __restrict__ out, long ldo,
                                int K, int ntiles_n)
{
  int wave = threadIdx.x >> 5;
  int tile = blockIdx.x * 8 + wave;
  int mt = tile / ntiles_n;
  int nt = tile - mt * ntiles_n;
  gemm_tile_f16wmma(A, lda, (long)mt * 16,
                    B, ldb, (long)nt * 16,
                    K, out, ldo, bias);
}

// ---------------- one-time data staging -----------------------------
// weights f32 [N,K] -> f16 [N,K]
__global__ void cvt_f16_kernel(const float* __restrict__ src,
                               _Float16* __restrict__ dst, int n)
{
  int i = blockIdx.x * blockDim.x + threadIdx.x;
  if (i < n) dst[i] = (_Float16)src[i];
}

// feats [c, m] f32 -> featsT16 [m, c] f16 (LDS-tiled 32x32 transpose)
__global__ void transpose_cvt_kernel(const float* __restrict__ feats,
                                     _Float16* __restrict__ ft)
{
  __shared__ float tile[32][33];
  int mBlk = blockIdx.x * 32;
  int cBlk = blockIdx.y * 32;
  int tx = threadIdx.x & 31;        // 256 threads = 32 x 8
  int ty = threadIdx.x >> 5;
#pragma unroll
  for (int i = 0; i < 32; i += 8) { // coalesced read along m
    tile[ty + i][tx] = feats[(long)(cBlk + ty + i) * MROWS + mBlk + tx];
  }
  __syncthreads();
#pragma unroll
  for (int i = 0; i < 32; i += 8) { // coalesced write along c
    ft[(long)(mBlk + ty + i) * NC + cBlk + tx] = (_Float16)tile[tx][ty + i];
  }
}

__global__ void zero_h_kernel(float* __restrict__ h, _Float16* __restrict__ h16)
{
  int i = blockIdx.x * blockDim.x + threadIdx.x;
  h[i] = 0.0f;
  h16[i] = (_Float16)0.0f;
}

// ---------------- score + softmax (one block per batch) -------------
__global__ void score_softmax_kernel(const float* __restrict__ fp,   // [32768,512]
                                     const float* __restrict__ hp,   // [128,512]
                                     const float* __restrict__ wsc,  // [512]
                                     float* __restrict__ alpha)      // [128,256]
{
  const int b = blockIdx.x;     // 0..127
  const int t = threadIdx.x;    // 0..255
  const float4* f4 = (const float4*)(fp + ((long)(b * NT + t)) * HID);
  const float4* h4 = (const float4*)(hp + (long)b * HID);
  const float4* w4 = (const float4*)wsc;
  float e = 0.0f;
#pragma unroll 4
  for (int i = 0; i < HID / 4; ++i) {
    float4 fv = f4[i], hv = h4[i], wv = w4[i];
    e += tanhf(fv.x + hv.x) * wv.x + tanhf(fv.y + hv.y) * wv.y +
         tanhf(fv.z + hv.z) * wv.z + tanhf(fv.w + hv.w) * wv.w;
  }
  __shared__ float sm[NT];
  sm[t] = e;
  __syncthreads();
  for (int off = NT / 2; off > 0; off >>= 1) {
    if (t < off) sm[t] = fmaxf(sm[t], sm[t + off]);
    __syncthreads();
  }
  float mx = sm[0];
  __syncthreads();
  float ex = __expf(e - mx);
  sm[t] = ex;
  __syncthreads();
  for (int off = NT / 2; off > 0; off >>= 1) {
    if (t < off) sm[t] += sm[t + off];
    __syncthreads();
  }
  alpha[(long)b * NT + t] = ex / sm[0];
}

// ---------------- context = einsum('cbt,bt->bc') (f16 out) ----------
__global__ void context_kernel(const float* __restrict__ feats,  // [512,128,256]
                               const float* __restrict__ alpha,  // [128,256]
                               _Float16* __restrict__ ctx16)     // [128,512]
{
  int idx = blockIdx.x * blockDim.x + threadIdx.x;   // 65536
  int b = idx >> 9;
  int c = idx & 511;
  const float4* f4 = (const float4*)(feats + (long)c * (NB * NT) + (long)b * NT);
  const float4* a4 = (const float4*)(alpha + (long)b * NT);
  float s = 0.0f;
#pragma unroll 4
  for (int i = 0; i < NT / 4; ++i) {
    float4 fv = f4[i], av = a4[i];
    s += fv.x * av.x + fv.y * av.y + fv.z * av.z + fv.w * av.w;
  }
  ctx16[idx] = (_Float16)s;
}

// ---------------- GRU gate fusion -----------------------------------
__device__ __forceinline__ float sigmoidf(float x) {
  return 1.0f / (1.0f + __expf(-x));
}

__global__ void gru_elem_kernel(const float* __restrict__ gi,      // [128,1536]
                                const float* __restrict__ gh,      // [128,1536]
                                float* __restrict__ h,             // [128,512]
                                _Float16* __restrict__ h16,        // [128,512]
                                _Float16* __restrict__ Hbuf16,     // [128,32,512]
                                int step)
{
  int idx = blockIdx.x * blockDim.x + threadIdx.x;  // 65536
  int b = idx >> 9;
  int j = idx & 511;
  const float* gib = gi + (long)b * (3 * HID);
  const float* ghb = gh + (long)b * (3 * HID);
  float r = sigmoidf(gib[j] + ghb[j]);
  float z = sigmoidf(gib[HID + j] + ghb[HID + j]);
  float n = tanhf(gib[2 * HID + j] + r * ghb[2 * HID + j]);
  float hv = h[idx];
  float hn = (1.0f - z) * n + z * hv;
  h[idx] = hn;
  h16[idx] = (_Float16)hn;
  Hbuf16[((long)b * STEPS + step) * HID + j] = (_Float16)hn;  // [nB, T, H] order
}

// ---------------- host-side orchestration ---------------------------
extern "C" void kernel_launch(void* const* d_in, const int* in_sizes, int n_in,
                              void* d_out, int out_size, void* d_ws, size_t ws_size,
                              hipStream_t stream)
{
  const float* feats  = (const float*)d_in[0];
  // d_in[1] text_length: all == STEPS, unused
  const float* Wi2h   = (const float*)d_in[2];   // [512,512]
  const float* Wh2h   = (const float*)d_in[3];   // [512,512]
  const float* bh2h   = (const float*)d_in[4];   // [512]
  const float* Wscore = (const float*)d_in[5];   // [1,512]
  const float* Wih    = (const float*)d_in[6];   // [1536,512]
  const float* Whh    = (const float*)d_in[7];   // [1536,512]
  const float* bih    = (const float*)d_in[8];   // [1536]
  const float* bhh    = (const float*)d_in[9];   // [1536]
  const float* Wgen   = (const float*)d_in[10];  // [96,512]
  const float* bgen   = (const float*)d_in[11];  // [96]
  float* out = (float*)d_out;                    // [4096, 96]

  // ---- workspace layout: f32 region then f16 region (~112 MB) ----
  float* wsf = (float*)d_ws;
  float* fp    = wsf;                        // feats_proj [32768,512]
  float* hp    = fp    + (long)MROWS * HID;  // [128,512]
  float* h     = hp    + NB * HID;           // [128,512]
  float* alpha = h     + NB * HID;           // [128,256]
  float* gi    = alpha + NB * NT;            // [128,1536]
  float* gh    = gi    + NB * 3 * HID;       // [128,1536]
  _Float16* wsh     = (_Float16*)(gh + NB * 3 * HID);
  _Float16* featsT16 = wsh;                          // [32768,512]
  _Float16* Wi2h16   = featsT16 + (long)MROWS * NC;  // [512,512]
  _Float16* Wh2h16   = Wi2h16   + HID * NC;          // [512,512]
  _Float16* Wih16    = Wh2h16   + HID * HID;         // [1536,512]
  _Float16* Whh16    = Wih16    + 3 * HID * NC;      // [1536,512]
  _Float16* Wgen16   = Whh16    + 3 * HID * HID;     // [96,512]
  _Float16* h16      = Wgen16   + NCLS1 * HID;       // [128,512]
  _Float16* ctx16    = h16      + NB * HID;          // [128,512]
  _Float16* Hbuf16   = ctx16    + NB * NC;           // [128,32,512]

  // ---- one-time staging ----
  cvt_f16_kernel<<<(HID * NC) / 256, 256, 0, stream>>>(Wi2h, Wi2h16, HID * NC);
  cvt_f16_kernel<<<(HID * HID) / 256, 256, 0, stream>>>(Wh2h, Wh2h16, HID * HID);
  cvt_f16_kernel<<<(3 * HID * NC) / 256, 256, 0, stream>>>(Wih, Wih16, 3 * HID * NC);
  cvt_f16_kernel<<<(3 * HID * HID) / 256, 256, 0, stream>>>(Whh, Whh16, 3 * HID * HID);
  cvt_f16_kernel<<<(NCLS1 * HID) / 256, 256, 0, stream>>>(Wgen, Wgen16, NCLS1 * HID);
  {
    dim3 g(MROWS / 32, NC / 32);
    transpose_cvt_kernel<<<g, 256, 0, stream>>>(feats, featsT16);
  }
  zero_h_kernel<<<(NB * HID) / 256, 256, 0, stream>>>(h, h16);

  // 1) feats_proj = featsT16 @ Wi2h16.T : tiles 2048 x 32 -> 8192 blocks
  gemm_f16_kernel<<<8192, 256, 0, stream>>>(featsT16, (long)NC, Wi2h16, (long)NC,
                                            nullptr, fp, (long)HID, NC, HID / 16);

  // 2) recurrence
  for (int s = 0; s < STEPS; ++s) {
    // hp = h @ Wh2h.T + bh2h : tiles 8x32 -> 32 blocks
    gemm_f16_kernel<<<32, 256, 0, stream>>>(h16, (long)HID, Wh2h16, (long)HID,
                                            bh2h, hp, (long)HID, HID, HID / 16);
    // alpha = softmax(tanh(fp + hp) @ w_score)
    score_softmax_kernel<<<NB, NT, 0, stream>>>(fp, hp, Wscore, alpha);
    // context (f16 out)
    context_kernel<<<(NB * NC) / 256, 256, 0, stream>>>(feats, alpha, ctx16);
    // gi = ctx @ Wih.T + bih : tiles 8x96 -> 96 blocks
    gemm_f16_kernel<<<96, 256, 0, stream>>>(ctx16, (long)NC, Wih16, (long)NC,
                                            bih, gi, (long)(3 * HID), NC, (3 * HID) / 16);
    // gh = h @ Whh.T + bhh
    gemm_f16_kernel<<<96, 256, 0, stream>>>(h16, (long)HID, Whh16, (long)HID,
                                            bhh, gh, (long)(3 * HID), HID, (3 * HID) / 16);
    // gates + h update + Hbuf16
    gru_elem_kernel<<<(NB * HID) / 256, 256, 0, stream>>>(gi, gh, h, h16, Hbuf16, s);
  }

  // 3) probs = Hbuf16[4096,512] @ Wgen16.T + bgen -> [4096,96]
  //    tiles: 256 x 6 = 1536 -> 192 blocks
  gemm_f16_kernel<<<192, 256, 0, stream>>>(Hbuf16, (long)HID, Wgen16, (long)HID,
                                           bgen, out, (long)NCLS1, HID, NCLS1 / 16);
}
